// GIN_27161373180011
// MI455X (gfx1250) — compile-verified, matrix operand order checked by reading, not run on hardware
//
#include <hip/hip_runtime.h>
#include <hip/hip_bf16.h>

#define N_NODES  32768
#define N_EDGES  524288
#define N_GRAPHS 512
#define F_INDIM  64
#define HDIM     128
#define NCLS     2
#define BN_EPS   1e-3f
#define KCHUNK   128   // K-chunk staged in LDS (64 cols x 128 k x bf16 = 16 KB)

typedef __attribute__((ext_vector_type(8)))  float          v8f;
typedef __attribute__((ext_vector_type(16))) __bf16         v16bf;
typedef __attribute__((ext_vector_type(8)))  unsigned short v8us;
typedef __attribute__((ext_vector_type(16))) unsigned short v16us;

__device__ __forceinline__ unsigned short f2bf(float x) {
  unsigned u = __float_as_uint(x);
  return (unsigned short)((u + 0x7FFFu + ((u >> 16) & 1u)) >> 16);  // RNE
}

__global__ void k_copy_f32(const float* __restrict__ s, float* __restrict__ d, int n) {
  int i = blockIdx.x * blockDim.x + threadIdx.x;
  if (i < n) d[i] = s[i];
}

__global__ void k_f32_to_bf16(const float* __restrict__ s, unsigned short* __restrict__ d, int n) {
  int i = blockIdx.x * blockDim.x + threadIdx.x;
  if (i < n) d[i] = f2bf(s[i]);
}

// W: K x Ncol row-major f32  ->  Wt: Ncol x K row-major bf16 (transposed for contiguous B-frag loads)
__global__ void k_wt_bf16(const float* __restrict__ W, unsigned short* __restrict__ Wt, int K, int Ncol) {
  int i = blockIdx.x * blockDim.x + threadIdx.x;
  if (i >= K * Ncol) return;
  int n = i / K, k = i - n * K;
  Wt[i] = f2bf(W[k * Ncol + n]);
}

// agg[src] += w * h[dst] ; one thread per (edge, 4 features)
__global__ void k_scatter(const float* __restrict__ h, const float* __restrict__ w,
                          const int* __restrict__ srcv, const int* __restrict__ dstv,
                          float* __restrict__ agg, int chunks, int din) {
  long long t = (long long)blockIdx.x * blockDim.x + threadIdx.x;
  long long total = (long long)N_EDGES * chunks;
  if (t >= total) return;
  int e  = (int)(t / chunks);
  int f4 = (int)(t - (long long)e * chunks) * 4;
  float wt = w[e];
  int d = dstv[e], s = srcv[e];
  const float4 hv = *(const float4*)(h + (long long)d * din + f4);
  float* base = agg + (long long)s * din + f4;
  atomicAdd(base + 0, wt * hv.x);
  atomicAdd(base + 1, wt * hv.y);
  atomicAdd(base + 2, wt * hv.z);
  atomicAdd(base + 3, wt * hv.w);
}

// segment-sum pooling into hg[:, colOff:colOff+H]
__global__ void k_pool(const float* __restrict__ h, const int* __restrict__ gid,
                       float* __restrict__ hg, int colOff, int ldout) {
  int t = blockIdx.x * blockDim.x + threadIdx.x;
  if (t >= N_NODES * HDIM) return;
  int i = t / HDIM, f = t - i * HDIM;
  atomicAdd(&hg[(long long)gid[i] * ldout + colOff + f], h[t]);
}

// WMMA GEMM: C = act(A[M x KTOT]bf16 @ B[KTOT x N]bf16 + bias), optional fused BN -> C2.
// KTOT is compile-time so all k-loops fully unroll (pipelined WMMA).
// One wave = one 16x16 tile; 4 waves/block share one tile_m and 4 consecutive tile_n;
// the block async-stages its 64-column weight slab (per 128-K chunk) into LDS
// (GLOBAL_LOAD_ASYNC_TO_LDS_B128 / ASYNCcnt), then feeds WMMA B-fragments from LDS.
// Grid is sized exactly (no guards) so EXEC is all-ones for WMMA and barriers.
template <int KTOT>
__global__ void k_wmma_gemm(const unsigned short* __restrict__ A,   // M x KTOT (row-major bf16)
                            const unsigned short* __restrict__ Bt,  // Ncol x KTOT (bf16, transposed)
                            const float* __restrict__ bias,
                            float* __restrict__ C,
                            float* __restrict__ C2,
                            const float* __restrict__ bng, const float* __restrict__ bnb,
                            const float* __restrict__ bnm, const float* __restrict__ bnv,
                            int Ncol, int do_relu) {
  constexpr int KC          = (KTOT < KCHUNK) ? KTOT : KCHUNK;
  constexpr int unitsPerRow = KC / 8;                  // 16B units per row-chunk
  constexpr int totalUnits  = 64 * unitsPerRow;        // multiple of 128 -> no divergence
  constexpr int NCHUNK      = KTOT / KC;

  __shared__ unsigned short smemB[64 * KC];            // <=16 KB slab: 64 cols x KC k
  const int tid    = threadIdx.x;
  const int lane   = tid & 31;
  const int wave   = tid >> 5;
  const int tilesN = Ncol >> 4;                        // 8 (divisible by 4)
  const int tile   = blockIdx.x * 4 + wave;
  const int tile_m = tile / tilesN;
  const int tile_n = tile - tile_m * tilesN;
  const int blk_n0 = (tile_n & ~3) * 16;               // block's first column (uniform in block)
  const int l15 = lane & 15;
  const int kb  = (lane < 16) ? 0 : 8;                 // ISA 16-bit A/B lane K-base
  const int row  = tile_m * 16 + l15;
  const int ncol = tile_n * 16 + l15;
  const int ncol_local = ncol - blk_n0;

  v8f acc = {};
#pragma unroll 1
  for (int c = 0; c < NCHUNK; ++c) {
    const int kc0 = c * KC;
    // ---- async-stage 64 x KC bf16 weight slab: global -> LDS (ASYNCcnt path) ----
#pragma unroll
    for (int u0 = 0; u0 < totalUnits / 128; ++u0) {
      const int u  = u0 * 128 + tid;
      const int nr = u / unitsPerRow;
      const int ke = (u - nr * unitsPerRow) * 8;       // element offset within chunk
      const unsigned short* gsrc = Bt + (size_t)(blk_n0 + nr) * KTOT + kc0 + ke;
      unsigned           lds = (unsigned)(size_t)(&smemB[nr * KC + ke]);
      unsigned long long ga  = (unsigned long long)(size_t)gsrc;
      asm volatile("global_load_async_to_lds_b128 %0, %1, off"
                   :: "v"(lds), "v"(ga) : "memory");
    }
    asm volatile("s_wait_asynccnt 0x0" ::: "memory");
    __syncthreads();

    const unsigned short* pa = A + (size_t)row * KTOT + kc0 + kb;
#pragma unroll
    for (int kk = 0; kk < KC; kk += 32) {
      v8us alo = *(const v8us*)(pa + kk);                               // global: K=kb+0..7
      v8us ahi = *(const v8us*)(pa + kk + 16);                          // global: K=kb+16..23
      v8us blo = *(const v8us*)(&smemB[ncol_local * KC + kk + kb]);     // LDS
      v8us bhi = *(const v8us*)(&smemB[ncol_local * KC + kk + kb + 16]);// LDS
      v16us au, bu;
#pragma unroll
      for (int i = 0; i < 8; ++i) { au[i] = alo[i]; au[i + 8] = ahi[i];
                                    bu[i] = blo[i]; bu[i + 8] = bhi[i]; }
      v16bf av  = __builtin_bit_cast(v16bf, au);
      v16bf bvv = __builtin_bit_cast(v16bf, bu);
      acc = __builtin_amdgcn_wmma_f32_16x16x32_bf16(false, av, false, bvv,
                                                    (short)0, acc, false, false);
    }
    __syncthreads();  // slab reuse barrier before next chunk
  }

  const float bsv = bias[ncol];
  float scale = 0.f, shift = 0.f;
  if (C2) { scale = rsqrtf(bnv[ncol] + BN_EPS) * bng[ncol];
            shift = bnb[ncol] - bnm[ncol] * scale; }
  const int mrow0 = tile_m * 16 + ((lane < 16) ? 0 : 8);  // acc VGPR j -> M = j (+8 hi lanes)
#pragma unroll
  for (int j = 0; j < 8; ++j) {
    float val = acc[j] + bsv;
    if (do_relu) val = fmaxf(val, 0.f);
    long long idx = (long long)(mrow0 + j) * Ncol + ncol;
    C[idx] = val;
    if (C2) C2[idx] = val * scale + shift;
  }
}

// tiny readout GEMM: out[G x C] = r[G x 128] @ Wm2[128 x C] + bm2
__global__ void k_final(const float* __restrict__ r, const float* __restrict__ Wm2,
                        const float* __restrict__ bm2, float* __restrict__ out) {
  int t = blockIdx.x * blockDim.x + threadIdx.x;
  if (t >= N_GRAPHS * NCLS) return;
  int g = t / NCLS, c = t - g * NCLS;
  float s = bm2[c];
  const float* rp = r + g * HDIM;
  for (int k = 0; k < HDIM; ++k) s += rp[k] * Wm2[k * NCLS + c];
  out[t] = s;
}

extern "C" void kernel_launch(void* const* d_in, const int* in_sizes, int n_in,
                              void* d_out, int out_size, void* d_ws, size_t ws_size,
                              hipStream_t stream) {
  (void)in_sizes; (void)n_in; (void)out_size; (void)ws_size;
  const float* x  = (const float*)d_in[0];
  const float* ew = (const float*)d_in[1];
  const float *W1[5], *b1[5], *W2[5], *b2[5];
  for (int l = 0; l < 5; ++l) {
    W1[l] = (const float*)d_in[2 + 4 * l]; b1[l] = (const float*)d_in[3 + 4 * l];
    W2[l] = (const float*)d_in[4 + 4 * l]; b2[l] = (const float*)d_in[5 + 4 * l];
  }
  const float *bg[3], *bb[3], *bm[3], *bv[3];
  for (int j = 0; j < 3; ++j) {
    bg[j] = (const float*)d_in[22 + 4 * j]; bb[j] = (const float*)d_in[23 + 4 * j];
    bm[j] = (const float*)d_in[24 + 4 * j]; bv[j] = (const float*)d_in[25 + 4 * j];
  }
  const float* Wm1 = (const float*)d_in[34];
  const float* bm1 = (const float*)d_in[35];
  const float* Wm2 = (const float*)d_in[36];
  const float* bm2 = (const float*)d_in[37];
  const int* esrc = (const int*)d_in[38];          // edge_index[0]
  const int* edst = esrc + N_EDGES;                // edge_index[1]
  const int* gid  = (const int*)d_in[39];
  float* out = (float*)d_out;

  // ---- workspace carve (256B aligned) ----
  char* ws = (char*)d_ws; size_t off = 0;
  auto carve = [&](size_t b) -> void* {
    void* p = ws + off; off = (off + b + 255) & ~(size_t)255; return p;
  };
  float*          A   = (float*)carve((size_t)N_NODES * HDIM * 4);       // bn'd activations
  float*          Bf  = (float*)carve((size_t)N_NODES * HDIM * 4);       // agg / mid / raw h
  unsigned short* X   = (unsigned short*)carve((size_t)N_NODES * HDIM * 2);
  float*          hg  = (float*)carve((size_t)N_GRAPHS * 5 * HDIM * 4);
  unsigned short* hgb = (unsigned short*)carve((size_t)N_GRAPHS * 5 * HDIM * 2);
  float*          rbuf= (float*)carve((size_t)N_GRAPHS * HDIM * 4);
  unsigned short* w1t[5]; unsigned short* w2t[5];
  w1t[0] = (unsigned short*)carve((size_t)HDIM * F_INDIM * 2);
  for (int l = 1; l < 5; ++l) w1t[l] = (unsigned short*)carve((size_t)HDIM * HDIM * 2);
  for (int l = 0; l < 5; ++l) w2t[l] = (unsigned short*)carve((size_t)HDIM * HDIM * 2);
  unsigned short* wm1t = (unsigned short*)carve((size_t)HDIM * 5 * HDIM * 2);

  const int TB = 256;
  // ---- weight prep: transpose + bf16 ----
  k_wt_bf16<<<(F_INDIM * HDIM + TB - 1) / TB, TB, 0, stream>>>(W1[0], w1t[0], F_INDIM, HDIM);
  for (int l = 1; l < 5; ++l)
    k_wt_bf16<<<(HDIM * HDIM + TB - 1) / TB, TB, 0, stream>>>(W1[l], w1t[l], HDIM, HDIM);
  for (int l = 0; l < 5; ++l)
    k_wt_bf16<<<(HDIM * HDIM + TB - 1) / TB, TB, 0, stream>>>(W2[l], w2t[l], HDIM, HDIM);
  k_wt_bf16<<<(5 * HDIM * HDIM + TB - 1) / TB, TB, 0, stream>>>(Wm1, wm1t, 5 * HDIM, HDIM);

  hipMemsetAsync(hg, 0, (size_t)N_GRAPHS * 5 * HDIM * 4, stream);

  const int bnsel[5] = {0, 0, 1, 2, 2};  // bn1,bn1,bn2,bn3,bn3 (as in source)
  const int tiles = (N_NODES / 16) * (HDIM / 16);   // 16384, exact multiple of 4

  for (int l = 0; l < 5; ++l) {
    const int din = (l == 0) ? F_INDIM : HDIM;
    const float* hin = (l == 0) ? x : A;
    // agg = hin + scatter(w * hin[dst] -> src)
    k_copy_f32<<<(N_NODES * din + TB - 1) / TB, TB, 0, stream>>>(hin, Bf, N_NODES * din);
    long long tot = (long long)N_EDGES * (din / 4);
    k_scatter<<<(unsigned)((tot + TB - 1) / TB), TB, 0, stream>>>(hin, ew, esrc, edst, Bf, din / 4, din);
    // bf16 convert -> GEMM1 (relu) -> Bf
    k_f32_to_bf16<<<(N_NODES * din + TB - 1) / TB, TB, 0, stream>>>(Bf, X, N_NODES * din);
    if (l == 0)
      k_wmma_gemm<F_INDIM><<<tiles / 4, 128, 0, stream>>>(X, w1t[0], b1[0], Bf, nullptr,
                                                          nullptr, nullptr, nullptr, nullptr,
                                                          HDIM, 1);
    else
      k_wmma_gemm<HDIM><<<tiles / 4, 128, 0, stream>>>(X, w1t[l], b1[l], Bf, nullptr,
                                                       nullptr, nullptr, nullptr, nullptr,
                                                       HDIM, 1);
    // bf16 convert -> GEMM2 (+fused BN): raw h -> Bf, bn(h) -> A
    k_f32_to_bf16<<<(N_NODES * HDIM + TB - 1) / TB, TB, 0, stream>>>(Bf, X, N_NODES * HDIM);
    const int j = bnsel[l];
    k_wmma_gemm<HDIM><<<tiles / 4, 128, 0, stream>>>(X, w2t[l], b2[l], Bf, A,
                                                     bg[j], bb[j], bm[j], bv[j],
                                                     HDIM, 0);
    // pool: layers 0..3 use raw h; layer 4 pools bn3(h5) per reference
    const float* psrc = (l == 4) ? A : Bf;
    k_pool<<<(N_NODES * HDIM + TB - 1) / TB, TB, 0, stream>>>(psrc, gid, hg, l * HDIM, 5 * HDIM);
  }

  // readout: relu(hg @ Wm1 + bm1) via WMMA (K=640 -> 5 LDS chunks), then tiny 128x2 GEMM
  k_f32_to_bf16<<<(N_GRAPHS * 5 * HDIM + TB - 1) / TB, TB, 0, stream>>>(hg, hgb, N_GRAPHS * 5 * HDIM);
  const int tilesR = (N_GRAPHS / 16) * (HDIM / 16);  // 256
  k_wmma_gemm<5 * HDIM><<<tilesR / 4, 128, 0, stream>>>(hgb, wm1t, bm1, rbuf, nullptr,
                                                        nullptr, nullptr, nullptr, nullptr,
                                                        HDIM, 1);
  k_final<<<(N_GRAPHS * NCLS + TB - 1) / TB, TB, 0, stream>>>(rbuf, Wm2, bm2, out);
}
